// GraphConvolution_86036784873686
// MI455X (gfx1250) — matrix operand order repaired and checked by reading, not compiled, not run on hardware
//
#include <hip/hip_runtime.h>

#define NNODE 20000
#define NPB   10                 // nodes per workgroup
#define NBLK  (NNODE / NPB)      // 2000 blocks
#define KNB   32
#define DIM   128
#define GATED 32
#define WTS   136                // padded LDS row stride (bf16) for Wt
#define XS    136                // padded LDS row stride (bf16) for x rows
#define HS    (DIM + 1)          // padded f32 row stride for h_nb

typedef __attribute__((ext_vector_type(16))) __bf16   v16bf;
typedef __attribute__((ext_vector_type(8)))  __bf16   bf16x8;
typedef __attribute__((ext_vector_type(4)))  __bf16   bf16x4;
typedef __attribute__((ext_vector_type(8)))  float    v8f;
typedef __attribute__((ext_vector_type(4)))  float    f32x4;
typedef __attribute__((ext_vector_type(4)))  unsigned u32x4;

// ---------------------------------------------------------------------------
// Prep: Wt[n][k] = (bf16) W[k][n]   (32 KB in d_ws; L2-resident forever after)
// ---------------------------------------------------------------------------
__global__ __launch_bounds__(256) void prep_wt(const float* __restrict__ W,
                                               __bf16* __restrict__ Wt) {
  int t = blockIdx.x * 256 + threadIdx.x;      // 0..16383
  int k = t >> 7, n = t & 127;
  Wt[n * DIM + k] = (__bf16)W[k * DIM + n];
}

// ---------------------------------------------------------------------------
// Main: one workgroup = NPB nodes, 256 threads = 8 wave32s.
// ---------------------------------------------------------------------------
__global__ __launch_bounds__(256) void gat_node(
    const float* __restrict__ x_self, const float* __restrict__ x_nb,
    const float* __restrict__ deg,    const float* __restrict__ bias,
    const float* __restrict__ phi,    const float* __restrict__ w1,
    const float* __restrict__ b1,     const float* __restrict__ w2,
    const float* __restrict__ b2,     const float* __restrict__ wself,
    const float* __restrict__ wnb,    const __bf16* __restrict__ Wt,
    float* __restrict__ out) {
  __shared__ __bf16 sWt[DIM * WTS];          // W^T bf16, padded rows (34816 B)
  __shared__ __bf16 sX [(KNB + 2) * XS];     // rows 0-31 nb, 32 self, 33 zero (9248 B)
  __shared__ float  sH [KNB * HS];           // h_nb f32 (16512 B)
  __shared__ float  sHs[DIM];                // h_self
  __shared__ float  sGate[KNB];
  __shared__ float  sWgt [KNB];              // att * gate
  __shared__ float  sPhiN[DIM];
  __shared__ float  sPhiS[DIM];
  __shared__ float  sG1[GATED];
  __shared__ float  sG2[GATED];

  const int t    = threadIdx.x;
  const int lane = t & 31;
  const int wave = t >> 5;
  const int g    = lane >> 4;          // K-half select
  const int lm   = lane & 15;
  const int col  = wave * 16 + lm;     // output column owned by this lane
  const int node0 = blockIdx.x * NPB;

  // ---- once-per-block staging: W^T (b128 copies, pad 128->136), constants ----
  {
    const u32x4* src = (const u32x4*)Wt;
    unsigned* dst = (unsigned*)sWt;
    #pragma unroll
    for (int i = 0; i < 8; ++i) {
      int j = t + 256 * i;                 // 16-byte chunk index 0..2047
      int n = j >> 4, kp = (j & 15) * 4;   // row, dword-in-row
      *(u32x4*)&dst[n * (WTS / 2) + kp] = src[j];
    }
    if (t < XS / 2) ((unsigned*)sX)[33 * (XS / 2) + t] = 0u;   // zero row
    if (t < DIM)                 { sPhiN[t] = phi[t]; sPhiS[t] = phi[DIM + t]; }
    else if (t < DIM + GATED)    { int j = t - DIM; sG1[j] = w1[j]; sG2[j] = w2[j]; }
  }
  const float bc  = bias[col];
  const float bb1 = b1[0], bb2 = b2[0];
  const float a_s = wself[0], a_n = wnb[0];

  for (int node = node0; node < node0 + NPB; ++node) {
    // ---- stage x_neighbor (32x128) + x_self row, f32 -> bf16, padded ----
    const float* xn = x_nb + (size_t)node * KNB * DIM;
    #pragma unroll
    for (int i = 0; i < 4; ++i) {
      int j = t + 256 * i;                 // float4 index 0..1023
      int e = j * 4;
      int r = e >> 7, c = e & 127;
      f32x4 v = *(const f32x4*)(xn + e);
      bf16x4 bv;
      bv[0] = (__bf16)v[0]; bv[1] = (__bf16)v[1];
      bv[2] = (__bf16)v[2]; bv[3] = (__bf16)v[3];
      *(bf16x4*)&sX[r * XS + c] = bv;
    }
    if (t < 32) {
      f32x4 v = *(const f32x4*)(x_self + (size_t)node * DIM + 4 * t);
      bf16x4 bv;
      bv[0] = (__bf16)v[0]; bv[1] = (__bf16)v[1];
      bv[2] = (__bf16)v[2]; bv[3] = (__bf16)v[3];
      *(bf16x4*)&sX[32 * XS + 4 * t] = bv;
    }
    // prefetch next node's neighbor tile into near caches (WGP scope) so it
    // streams HBM->WGP while this node's WMMA runs
    if (node + 1 < node0 + NPB) {
      const float* xn2 = xn + KNB * DIM;
      #pragma unroll
      for (int i = 0; i < 4; ++i)
        __builtin_prefetch(xn2 + (t + 256 * i) * 4, 0, 3);
    }
    __syncthreads();

    // ---- WMMA: [nb0-15 | nb16-31 | self] x W, bf16 in / f32 acc ----
    {
      v8f acc0 = {}; v8f acc1 = {}; v8f acc2 = {};
      const int r2 = (lm == 0) ? 32 : 33;  // self row for lm==0, zero row else
      #pragma unroll
      for (int s = 0; s < 4; ++s) {
        const int kk = s * 32;
        // B fragment: column `col`, K = kk + 16*g + (0..15) contiguous
        v16bf bf = *(const v16bf*)&sWt[col * WTS + kk + g * 16];
        // A fragments: row = lm (+tile), K = kk + 16*g + (0..15) contiguous
        union { v16bf v; bf16x8 h[2]; } a0, a1, a2;
        a0.h[0] = *(const bf16x8*)&sX[(lm     ) * XS + kk + g * 16];
        a0.h[1] = *(const bf16x8*)&sX[(lm     ) * XS + kk + g * 16 + 8];
        a1.h[0] = *(const bf16x8*)&sX[(lm + 16) * XS + kk + g * 16];
        a1.h[1] = *(const bf16x8*)&sX[(lm + 16) * XS + kk + g * 16 + 8];
        a2.h[0] = *(const bf16x8*)&sX[(r2     ) * XS + kk + g * 16];
        a2.h[1] = *(const bf16x8*)&sX[(r2     ) * XS + kk + g * 16 + 8];
        acc0 = __builtin_amdgcn_wmma_f32_16x16x32_bf16(false, a0.v, false, bf,
                                                       (short)0, acc0, false, false);
        acc1 = __builtin_amdgcn_wmma_f32_16x16x32_bf16(false, a1.v, false, bf,
                                                       (short)0, acc1, false, false);
        acc2 = __builtin_amdgcn_wmma_f32_16x16x32_bf16(false, a2.v, false, bf,
                                                       (short)0, acc2, false, false);
      }
      // D layout: element r -> row (r + 8*g) within tile, col = lm
      #pragma unroll
      for (int r = 0; r < 8; ++r) {
        sH[(g * 8 + r) * HS + col]      = acc0[r] + bc;
        sH[(16 + g * 8 + r) * HS + col] = acc1[r] + bc;
      }
      if (g == 0) sHs[col] = acc2[0] + bc;   // self row lives in lm==0's tile row 0
    }
    // ---- gate MLP (threads 128..159) ----
    if (t >= 128 && t < 128 + KNB) {
      const int k = t - 128;
      const float d = deg[(size_t)node * KNB + k];
      float acc = 0.f;
      #pragma unroll
      for (int j = 0; j < GATED; ++j) {
        float s = 1.f / (1.f + __expf(-fmaf(d, sG1[j], bb1)));
        acc = fmaf(s, sG2[j], acc);
      }
      sGate[k] = 1.f / (1.f + __expf(-(acc + bb2)));
    }
    __syncthreads();

    // ---- attention scores + softmax over K (wave 0, lane = neighbor) ----
    if (wave == 0) {
      float sc = 0.f, ss = 0.f;
      for (int i = 0; i < DIM; ++i) {
        sc = fmaf(sH[lane * HS + i], sPhiN[i], sc);
        ss = fmaf(sHs[i], sPhiS[i], ss);
      }
      float sco = sc + ss;
      sco = (sco > 0.f) ? sco : 0.2f * sco;           // leaky_relu(0.2)
      float m = sco;
      #pragma unroll
      for (int off = 16; off > 0; off >>= 1) m = fmaxf(m, __shfl_xor(m, off, 32));
      float e = __expf(sco - m);
      float sum = e;
      #pragma unroll
      for (int off = 16; off > 0; off >>= 1) sum += __shfl_xor(sum, off, 32);
      sWgt[lane] = (e / sum) * sGate[lane];
    }
    __syncthreads();

    // ---- weighted neighbor reduction + self mix + ReLU ----
    if (t < DIM) {
      float acc = 0.f;
      #pragma unroll
      for (int k = 0; k < KNB; ++k) acc = fmaf(sWgt[k], sH[k * HS + t], acc);
      float o = a_s * sHs[t] + a_n * acc;
      out[(size_t)node * DIM + t] = fmaxf(o, 0.f);
    }
    __syncthreads();   // protect sX/sH reuse before next node's staging
  }
}

// ---------------------------------------------------------------------------
extern "C" void kernel_launch(void* const* d_in, const int* in_sizes, int n_in,
                              void* d_out, int out_size, void* d_ws, size_t ws_size,
                              hipStream_t stream) {
  const float* x_self = (const float*)d_in[0];
  const float* x_nb   = (const float*)d_in[1];
  const float* deg    = (const float*)d_in[2];
  const float* W      = (const float*)d_in[3];
  const float* bias   = (const float*)d_in[4];
  const float* phi    = (const float*)d_in[5];
  const float* w1     = (const float*)d_in[6];
  const float* b1     = (const float*)d_in[7];
  const float* w2     = (const float*)d_in[8];
  const float* b2     = (const float*)d_in[9];
  const float* wslf   = (const float*)d_in[10];
  const float* wnbr   = (const float*)d_in[11];
  float* outp = (float*)d_out;
  __bf16* Wt  = (__bf16*)d_ws;   // 32 KB scratch

  prep_wt<<<64, 256, 0, stream>>>(W, Wt);
  gat_node<<<NBLK, 256, 0, stream>>>(x_self, x_nb, deg, bias, phi,
                                     w1, b1, w2, b2, wslf, wnbr, Wt, outp);
}
